// Contraction_17600775979378
// MI455X (gfx1250) — matrix operand order repaired and checked by reading, not compiled, not run on hardware
//
#include <hip/hip_runtime.h>
#include <hip/hip_bf16.h>

typedef __attribute__((ext_vector_type(16))) __bf16 v16bf;
typedef __attribute__((ext_vector_type(8)))  __bf16 v8bf;
typedef __attribute__((ext_vector_type(8)))  float  v8f;

#define NPATH   34
#define NUM_C   64
#define NUM_H   64
#define NNODES  2048
#define NEDGES  8192
#define CG_N    3436
#define Z_N     738
#define ACC_TOTAL 20447232   // floats: 4*64*2048 + 9*64*2048*3 + 11*64*2048*5 + 10*64*2048*7

// ---------------- compile-time path metadata ----------------
struct Meta {
  int L1[NPATH], L2[NPATH], L3[NPATH];
  int CGO[NPATH];   // offset into CG table
  int ZO[NPATH];    // offset into per-edge z scratch
  int SLOT[NPATH];  // channel-block slot within l3 accumulator
  int CNT[4];
  constexpr Meta() : L1{}, L2{}, L3{}, CGO{}, ZO{}, SLOT{}, CNT{} {
    int idx = 0, cg = 0, z = 0;
    for (int a = 0; a <= 3; ++a)
      for (int b = 0; b <= 3; ++b) {
        int lo = a - b; if (lo < 0) lo = -lo;
        int hi = a + b; if (hi > 3) hi = 3;
        for (int c = lo; c <= hi; ++c) {
          L1[idx] = a; L2[idx] = b; L3[idx] = c;
          CGO[idx] = cg; cg += (2*a+1)*(2*b+1)*(2*c+1);
          ZO[idx]  = z;  z  += (2*a+1)*(2*c+1);
          SLOT[idx] = CNT[c]; CNT[c]++;
          ++idx;
        }
      }
  }
};
constexpr Meta META{};

// Deterministic stand-in CG coefficients (~N(0,0.2) scale), compile-time LCG.
struct CGTab {
  float v[CG_N];
  constexpr CGTab() : v{} {
    unsigned s = 0x9E3779B9u;
    for (int i = 0; i < CG_N; ++i) {
      s = s * 1664525u + 1013904223u;
      v[i] = 0.35f * (2.0f * (float)(s >> 8) / 16777216.0f - 1.0f);
    }
  }
};
constexpr CGTab CGT{};

// ---------------- kernel 1: zero accumulators ----------------
__global__ void zero_kernel(float* __restrict__ p, int n) {
  int i = blockIdx.x * blockDim.x + threadIdx.x;
  int stride = gridDim.x * blockDim.x;
  for (; i < n; i += stride) p[i] = 0.0f;
}

// ---------------- kernel 2: per-edge tensor product + scatter ----------------
__global__ void __launch_bounds__(64)
edge_tp_kernel(const float* __restrict__ x0, const float* __restrict__ x1,
               const float* __restrict__ x2, const float* __restrict__ x3,
               const float* __restrict__ y0, const float* __restrict__ y1,
               const float* __restrict__ y2, const float* __restrict__ y3,
               const float* __restrict__ ws, const int* __restrict__ ei,
               float* __restrict__ acc0, float* __restrict__ acc1,
               float* __restrict__ acc2, float* __restrict__ acc3)
{
  const int e = blockIdx.x;
  const int c = threadIdx.x;      // channel 0..63

  __shared__ float ysh[16];
  __shared__ float zsh[Z_N];
  __shared__ int   sd[2];

  if (c < 2) sd[c] = ei[c * NEDGES + e];      // src, dst
  if (c < 16) {
    float v;
    if (c == 0)      v = y0[e];
    else if (c < 4)  v = y1[e * 3 + (c - 1)];
    else if (c < 9)  v = y2[e * 5 + (c - 4)];
    else             v = y3[e * 7 + (c - 9)];
    ysh[c] = v;
  }
  __syncthreads();

  // z[l,n] = sum_m CG[l,m,n] * y[m]  (one path per thread; shared across channels)
  if (c < NPATH) {
    const int i  = c;
    const int d1 = 2 * META.L1[i] + 1;
    const int d2 = 2 * META.L2[i] + 1;
    const int d3 = 2 * META.L3[i] + 1;
    const float* cg = &CGT.v[META.CGO[i]];
    float* z = &zsh[META.ZO[i]];
    const int yb = META.L2[i] * META.L2[i];   // {0,1,4,9}
    for (int l = 0; l < d1; ++l)
      for (int n = 0; n < d3; ++n) {
        float s = 0.0f;
        for (int m = 0; m < d2; ++m) s += cg[(l * d2 + m) * d3 + n] * ysh[yb + m];
        z[l * d3 + n] = s;
      }
  }

  const int src = sd[0], dst = sd[1];
  // this channel's gathered source features, all degrees packed (1+3+5+7 = 16)
  float xr[16];
  xr[0] = x0[src * NUM_C + c];
  #pragma unroll
  for (int j = 0; j < 3; ++j) xr[1 + j] = x1[(src * NUM_C + c) * 3 + j];
  #pragma unroll
  for (int j = 0; j < 5; ++j) xr[4 + j] = x2[(src * NUM_C + c) * 5 + j];
  #pragma unroll
  for (int j = 0; j < 7; ++j) xr[9 + j] = x3[(src * NUM_C + c) * 7 + j];
  __syncthreads();

  float* const accs[4] = {acc0, acc1, acc2, acc3};
  const int Ks[4] = {256, 576, 704, 640};

  #pragma unroll
  for (int i = 0; i < NPATH; ++i) {
    const int l1 = META.L1[i], l3 = META.L3[i];
    const int d1 = 2 * l1 + 1, d3 = 2 * l3 + 1;
    const float* z  = &zsh[META.ZO[i]];
    const float* xl = &xr[l1 * l1];
    const float  w  = ws[(size_t)e * (NUM_C * NPATH) + i * NUM_C + c];
    float* acc = accs[l3];
    const int K    = Ks[l3];
    const int base = dst * K + META.SLOT[i] * NUM_C + c;
    const int NK   = NNODES * K;
    for (int n = 0; n < d3; ++n) {
      float s = 0.0f;
      for (int l = 0; l < d1; ++l) s += z[l * d3 + n] * xl[l];
      atomicAdd(&acc[n * NK + base], w * s);
    }
  }
}

// ---------------- kernel 3: per-l3 channel-mixing GEMM via bf16 WMMA ----------------
// out[n,h,m] = sum_k A[m][n][k] * W[k][h];  M=2048 rows, N=H=64 cols, K in {256,576,704,640}
// block = 256 threads (8 waves); wave w owns rows [blk*128 + w*16, +16) and ALL 4 column
// tiles (A-fragment reuse x4 -> 4 back-to-back WMMA per K-step per wave).
//
// W chunk pipeline per 32-K step:
//   GLOBAL_LOAD_ASYNC_TO_LDS_B128 (fp32, double-buffered) -> s_wait_asynccnt/barrier
//   -> cooperative fp32->bf16 convert into a TRANSPOSED fragment-ready LDS layout
//   -> B fragments are pure ds_load_b128 pairs (no VALU writes WMMA sources => no hazard nops)
//   -> next chunk's async copy is issued after the mid-barrier and overlaps the WMMA burst.
__device__ __forceinline__ void stage_async(const float* __restrict__ W, int k0,
                                            unsigned ldsBase, int tid) {
  const float* gsrc = W + (size_t)k0 * NUM_H + tid * 8;
  const unsigned l0 = ldsBase + (unsigned)tid * 32u;
  asm volatile(
    "global_load_async_to_lds_b128 %0, %1, off\n\t"
    "global_load_async_to_lds_b128 %2, %3, off"
    :: "v"(l0), "v"((unsigned long long)(uintptr_t)gsrc),
       "v"(l0 + 16u), "v"((unsigned long long)(uintptr_t)(gsrc + 4))
    : "memory");
}

#define BTR_W 5   // 5 x v8bf = 40 bf16 per column row (80B: 16B-aligned, bank-spread)

__global__ void __launch_bounds__(256)
linear_wmma_kernel(const float* __restrict__ A,   // [dm][2048][K] fp32 accumulators
                   const float* __restrict__ W,   // [K][64] fp32 weights
                   float* __restrict__ out,       // [2048][64][dm] fp32
                   int K, int dm)
{
  const int m    = blockIdx.y;
  const int tid  = threadIdx.x;
  const int wave = tid >> 5;
  const int lane = tid & 31;
  const int row0 = blockIdx.x * 128 + wave * 16;
  const int lo   = lane & 15;
  const bool hiHalf = lane >= 16;

  __shared__ float bsh[2][32 * NUM_H];        // double-buffered fp32 W chunks (async dest)
  __shared__ v8bf  btr[2][NUM_H][BTR_W];      // bf16, transposed: [buf][col][k/8]
  const unsigned base0 = (unsigned)(unsigned long long)(uintptr_t)(void*)&bsh[0][0];
  const unsigned base1 = (unsigned)(unsigned long long)(uintptr_t)(void*)&bsh[1][0];

  const float* Am   = A + (size_t)m * ((size_t)NNODES * (size_t)K);
  const float* arow = Am + (size_t)(row0 + lo) * K;

  // convert-pass coordinates: 4 threads per column, 8 k-values per thread
  const int ccol = tid & 63;
  const int ckk  = (tid >> 6) * 8;

  v8f c0 = {}, c1 = {}, c2 = {}, c3 = {};

  const int nsteps = K >> 5;                  // K is a multiple of 32
  stage_async(W, 0, base0, tid);              // prologue: fill fp32 buffer 0

  for (int it = 0; it < nsteps; ++it) {
    const int k0  = it << 5;
    const int buf = it & 1;

    asm volatile("s_wait_asynccnt 0" ::: "memory");   // this wave's chunk-`it` copy done
    __syncthreads();                          // chunk `it` visible; prior btr reads done

    // --- cooperative fp32 -> bf16 transpose into fragment-ready layout ---
    {
      const float* cur = bsh[buf];
      v8bf pk;
      #pragma unroll
      for (int j = 0; j < 8; ++j)
        pk[j] = (__bf16)cur[(ckk + j) * NUM_H + ccol];
      btr[buf][ccol][ckk >> 3] = pk;          // ds_store_b128
    }
    __syncthreads();                          // btr[buf] ready; fp32[buf] fully consumed

    if (it + 1 < nsteps)                      // overlap next copy with the WMMA burst
      stage_async(W, k0 + 32, buf ? base0 : base1, tid);

    // --- A fragment (direct global, rows contiguous in K) ---
    // lanes 0..15: K k0+0..7 | k0+16..23 ; lanes 16..31: K k0+8..15 | k0+24..31
    v16bf a;
    const int ka = k0 + (hiHalf ? 8 : 0);
    #pragma unroll
    for (int j = 0; j < 8; ++j) {
      a[j]     = (__bf16)arow[ka + j];
      a[8 + j] = (__bf16)arow[ka + 16 + j];
    }

    // --- B fragments: two ds_load_b128 per tile, no VALU into WMMA sources ---
    const int e0 = hiHalf ? 1 : 0;            // lanes 0..15: k 0..7,16..23; lanes 16..31: k 8..15,24..31
    const v8bf* bc0 = btr[buf][lo];
    const v8bf* bc1 = btr[buf][16 + lo];
    const v8bf* bc2 = btr[buf][32 + lo];
    const v8bf* bc3 = btr[buf][48 + lo];
    v16bf b0 = __builtin_shufflevector(bc0[e0], bc0[e0 + 2],
                 0,1,2,3,4,5,6,7,8,9,10,11,12,13,14,15);
    v16bf b1 = __builtin_shufflevector(bc1[e0], bc1[e0 + 2],
                 0,1,2,3,4,5,6,7,8,9,10,11,12,13,14,15);
    v16bf b2 = __builtin_shufflevector(bc2[e0], bc2[e0 + 2],
                 0,1,2,3,4,5,6,7,8,9,10,11,12,13,14,15);
    v16bf b3 = __builtin_shufflevector(bc3[e0], bc3[e0 + 2],
                 0,1,2,3,4,5,6,7,8,9,10,11,12,13,14,15);

    c0 = __builtin_amdgcn_wmma_f32_16x16x32_bf16(false, a, false, b0, (short)0, c0, false, false);
    c1 = __builtin_amdgcn_wmma_f32_16x16x32_bf16(false, a, false, b1, (short)0, c1, false, false);
    c2 = __builtin_amdgcn_wmma_f32_16x16x32_bf16(false, a, false, b2, (short)0, c2, false, false);
    c3 = __builtin_amdgcn_wmma_f32_16x16x32_bf16(false, a, false, b3, (short)0, c3, false, false);
  }

  // D layout: VGPR r -> row0 + r (lanes 0..15) / row0 + 8 + r (lanes 16..31)
  const int rbase = row0 + (hiHalf ? 8 : 0);
  #pragma unroll
  for (int t4 = 0; t4 < 4; ++t4) {
    const int col = t4 * 16 + lo;
    const v8f& cc = (t4 == 0) ? c0 : (t4 == 1) ? c1 : (t4 == 2) ? c2 : c3;
    #pragma unroll
    for (int r = 0; r < 8; ++r) {
      const int row = rbase + r;
      out[((size_t)row * NUM_H + col) * dm + m] = cc[r];
    }
  }
}

// ---------------- host launcher ----------------
extern "C" void kernel_launch(void* const* d_in, const int* in_sizes, int n_in,
                              void* d_out, int out_size, void* d_ws, size_t ws_size,
                              hipStream_t stream) {
  (void)in_sizes; (void)n_in; (void)out_size; (void)ws_size;
  // setup_inputs() dict order: x0,y0,w_lin0, x1,y1,w_lin1, x2,y2,w_lin2, x3,y3,w_lin3, ws, edge_index
  const float* x0  = (const float*)d_in[0];
  const float* y0  = (const float*)d_in[1];
  const float* wl0 = (const float*)d_in[2];
  const float* x1  = (const float*)d_in[3];
  const float* y1  = (const float*)d_in[4];
  const float* wl1 = (const float*)d_in[5];
  const float* x2  = (const float*)d_in[6];
  const float* y2  = (const float*)d_in[7];
  const float* wl2 = (const float*)d_in[8];
  const float* x3  = (const float*)d_in[9];
  const float* y3  = (const float*)d_in[10];
  const float* wl3 = (const float*)d_in[11];
  const float* ws  = (const float*)d_in[12];
  const int*   ei  = (const int*)d_in[13];
  float* out = (float*)d_out;

  // workspace: fp32 node accumulators per l3 (K = C*count)
  float* acc  = (float*)d_ws;
  float* acc0 = acc;                // 4*64*2048*1  =   524288
  float* acc1 = acc + 524288;       // 9*64*2048*3  =  3538944
  float* acc2 = acc + 4063232;      // 11*64*2048*5 =  7208960
  float* acc3 = acc + 11272192;     // 10*64*2048*7 =  9175040

  zero_kernel<<<dim3(2048), dim3(256), 0, stream>>>(acc, ACC_TOTAL);

  edge_tp_kernel<<<dim3(NEDGES), dim3(64), 0, stream>>>(
      x0, x1, x2, x3, y0, y1, y2, y3, ws, ei, acc0, acc1, acc2, acc3);

  // per-l3 linears: 2048/128 = 16 row-blocks; out bases in floats: 0, 131072, 524288, 1179648
  linear_wmma_kernel<<<dim3(16, 1), dim3(256), 0, stream>>>(acc0, wl0, out,            256, 1);
  linear_wmma_kernel<<<dim3(16, 3), dim3(256), 0, stream>>>(acc1, wl1, out + 131072,   576, 3);
  linear_wmma_kernel<<<dim3(16, 5), dim3(256), 0, stream>>>(acc2, wl2, out + 524288,   704, 5);
  linear_wmma_kernel<<<dim3(16, 7), dim3(256), 0, stream>>>(acc3, wl3, out + 1179648,  640, 7);
}
